// Attention_25769803776075
// MI455X (gfx1250) — matrix operand order repaired and checked by reading, not compiled, not run on hardware
//
#include <hip/hip_runtime.h>

typedef __attribute__((ext_vector_type(16))) __bf16 v16bf;
typedef __attribute__((ext_vector_type(8)))  __bf16 v8bf;
typedef __attribute__((ext_vector_type(8)))  float  v8f;

constexpr int BATCH = 4;
constexpr int SEQ   = 1024;
constexpr int DIM   = 4096;
constexpr int NH    = 32;
constexpr int HD    = 128;

__device__ __forceinline__ v16bf cat16(v8bf lo, v8bf hi) {
  return __builtin_shufflevector(lo, hi, 0,1,2,3,4,5,6,7,8,9,10,11,12,13,14,15);
}
// A-fragment: per-lane K elements {kb..kb+15} (p already at +kb)
__device__ __forceinline__ v16bf ldA(const __bf16* p) {
  v8bf lo = *(const v8bf*)p;
  v8bf hi = *(const v8bf*)(p + 8);
  return cat16(lo, hi);
}
// B-fragment: 16 contiguous K per lane
__device__ __forceinline__ v16bf ldB(const __bf16* p) {
  v8bf lo = *(const v8bf*)p;
  v8bf hi = *(const v8bf*)(p + 8);
  return cat16(lo, hi);
}
__device__ __forceinline__ v8f wmma_bf16(v16bf a, v16bf b, v8f c) {
  return __builtin_amdgcn_wmma_f32_16x16x32_bf16(false, a, false, b, (short)0, c, false, false);
}

// ---------------------------------------------------------------- convert ---
__global__ void cvt_f32_bf16(const float* __restrict__ src, __bf16* __restrict__ dst, int n) {
  int i = blockIdx.x * blockDim.x + threadIdx.x;
  int stride = gridDim.x * blockDim.x;
  for (; i < n; i += stride) dst[i] = (__bf16)src[i];
}

// ------------------------------------------------------------------- GEMM ---
// out[m,n] = sum_k A[m,k] * W[n,k].  M = N = K = 4096.
// Wave tile 64x64 (16 WMMAs / 16 b128 loads per k-step), WG = 8 waves (2M x 4N)
// covering a 128x256 macro tile.
// EPI 0: V  -> bf16, transposed to (B,H,HD,S)
// EPI 1: Q/K-> RoPE fused, bf16 to (B,H,S,HD)
// EPI 2: out-> fp32 to (B,S,DIM)
template<int EPI>
__global__ __launch_bounds__(256) void gemm_bf16(
    const __bf16* __restrict__ Am, const __bf16* __restrict__ Wm,
    void* __restrict__ outp,
    const float* __restrict__ cosp, const float* __restrict__ sinp)
{
  const int lane = threadIdx.x & 31;
  const int w    = threadIdx.x >> 5;      // 0..7
  const int wm   = w >> 2;                // 0..1  (M)
  const int wn   = w & 3;                 // 0..3  (N)
  const int hl   = lane >> 4;             // half-wave
  const int l15  = lane & 15;
  const int mBase = blockIdx.y * 128 + wm * 64;
  const int nBase = blockIdx.x * 256 + wn * 64;

  v8f acc[4][4];
#pragma unroll
  for (int t = 0; t < 4; t++)
#pragma unroll
    for (int u = 0; u < 4; u++) acc[t][u] = 0.f;

  const __bf16* aPtr[4];
  const __bf16* bPtr[4];
#pragma unroll
  for (int t = 0; t < 4; t++)
    aPtr[t] = Am + (size_t)(mBase + t * 16 + l15) * DIM + hl * 16;
#pragma unroll
  for (int u = 0; u < 4; u++)
    bPtr[u] = Wm + (size_t)(nBase + u * 16 + l15) * DIM + hl * 16;

  for (int k = 0; k < DIM; k += 32) {
    v16bf af[4], bfg[4];
#pragma unroll
    for (int t = 0; t < 4; t++) af[t] = ldA(aPtr[t] + k);
#pragma unroll
    for (int u = 0; u < 4; u++) bfg[u] = ldB(bPtr[u] + k);
#pragma unroll
    for (int t = 0; t < 4; t++)
#pragma unroll
      for (int u = 0; u < 4; u++)
        acc[t][u] = wmma_bf16(af[t], bfg[u], acc[t][u]);
  }

#pragma unroll
  for (int t = 0; t < 4; t++) {
#pragma unroll
    for (int u = 0; u < 4; u++) {
#pragma unroll
      for (int e = 0; e < 8; e++) {
        const int m = mBase + t * 16 + e + hl * 8;
        const int n = nBase + u * 16 + l15;
        const float v = acc[t][u][e];
        if constexpr (EPI == 2) {
          ((float*)outp)[(size_t)m * DIM + n] = v;
        } else if constexpr (EPI == 0) {
          const int h = n >> 7, d = n & 127;
          const int b = m >> 10, s = m & 1023;
          ((__bf16*)outp)[(((size_t)(b * NH + h) * HD + d) * SEQ) + s] = (__bf16)v;
        } else { // RoPE
          const float vp = __shfl_xor(v, 1, 32);
          const int h = n >> 7, d = n & 127;
          const int b = m >> 10, s = m & 1023;
          const int fi = d >> 1;
          const float c  = cosp[s * (HD / 2) + fi];
          const float si = sinp[s * (HD / 2) + fi];
          const float r = ((n & 1) == 0) ? (v * c - vp * si) : (vp * si + v * c);
          ((__bf16*)outp)[(((size_t)(b * NH + h) * SEQ + s) * HD) + d] = (__bf16)r;
        }
      }
    }
  }
}

// -------------------------------------------------------------- attention ---
// Q,K: (B,H,S,HD) bf16 (RoPE'd).  Vt: (B,H,HD,S) bf16.  attn: (B,S,DIM) bf16.
// Grid: x = S/128 (8), y = B*H (128). Wave w handles 16 query rows.
__global__ __launch_bounds__(256) void attn_flash(
    const __bf16* __restrict__ Qb, const __bf16* __restrict__ Kb,
    const __bf16* __restrict__ Vt, __bf16* __restrict__ attn)
{
  __shared__ __bf16 psc[8][16][32];   // per-wave P transpose scratch (8 KB)
  const int lane = threadIdx.x & 31;
  const int w    = threadIdx.x >> 5;
  const int hl   = lane >> 4;
  const int l15  = lane & 15;
  const int bh   = blockIdx.y;              // 0..127
  const int b    = bh >> 5;
  const int h    = bh & 31;
  const int q0   = blockIdx.x * 128 + w * 16;
  const float scale = 0.08838834764831845f; // 1/sqrt(128)

  // resident Q A-fragments covering HD=128
  const __bf16* Qrow = Qb + ((size_t)bh * SEQ + q0) * HD;
  v16bf qf[4];
#pragma unroll
  for (int f = 0; f < 4; f++)
    qf[f] = ldA(Qrow + (size_t)l15 * HD + f * 32 + hl * 16);

  v8f o[8];
#pragma unroll
  for (int t = 0; t < 8; t++) o[t] = 0.f;
  float mrow[8], lrow[8], alpha[8];
#pragma unroll
  for (int e = 0; e < 8; e++) { mrow[e] = -__builtin_inff(); lrow[e] = 0.f; }

  const __bf16* Kbase = Kb + (size_t)bh * SEQ * HD;
  const __bf16* Vbase = Vt + (size_t)bh * HD * SEQ;

  const int nChunks = (q0 + 15) / 32 + 1;  // causal: keys <= q0+15
  for (int c = 0; c < nChunks; c++) {
    const int k0 = c * 32;
    // ---- scores: two 16-key tiles, K-dim = HD in 4 steps of 32
    v8f s0 = 0.f, s1 = 0.f;
#pragma unroll
    for (int f = 0; f < 4; f++) {
      v16bf kb0 = ldB(Kbase + (size_t)(k0 + l15) * HD + f * 32 + hl * 16);
      v16bf kb1 = ldB(Kbase + (size_t)(k0 + 16 + l15) * HD + f * 32 + hl * 16);
      s0 = wmma_bf16(qf[f], kb0, s0);
      s1 = wmma_bf16(qf[f], kb1, s1);
    }
    // ---- online softmax (row reductions within 16-lane halves)
    const int key0 = k0 + l15, key1 = key0 + 16;
#pragma unroll
    for (int e = 0; e < 8; e++) {
      const int q = q0 + e + hl * 8;
      float a0 = s0[e] * scale + (key0 <= q ? 0.f : -__builtin_inff());
      float a1 = s1[e] * scale + (key1 <= q ? 0.f : -__builtin_inff());
      float cm = fmaxf(a0, a1);
      cm = fmaxf(cm, __shfl_xor(cm, 1, 32));
      cm = fmaxf(cm, __shfl_xor(cm, 2, 32));
      cm = fmaxf(cm, __shfl_xor(cm, 4, 32));
      cm = fmaxf(cm, __shfl_xor(cm, 8, 32));
      const float mn = fmaxf(mrow[e], cm);
      const float p0 = __expf(a0 - mn);
      const float p1 = __expf(a1 - mn);
      float rs = p0 + p1;
      rs += __shfl_xor(rs, 1, 32);
      rs += __shfl_xor(rs, 2, 32);
      rs += __shfl_xor(rs, 4, 32);
      rs += __shfl_xor(rs, 8, 32);
      alpha[e] = __expf(mrow[e] - mn);
      lrow[e]  = lrow[e] * alpha[e] + rs;
      mrow[e]  = mn;
      // C-layout -> LDS (row = query, col = key-in-chunk)
      psc[w][e + hl * 8][l15]      = (__bf16)p0;
      psc[w][e + hl * 8][16 + l15] = (__bf16)p1;
    }
    // ---- P as A-fragment (16x32) from LDS; DS ops are in-order per wave
    v16bf pf = ldA(&psc[w][l15][hl * 16]);
    // ---- O = O*alpha + P @ V   (8 d-tiles of 16)
#pragma unroll
    for (int t = 0; t < 8; t++) {
#pragma unroll
      for (int e = 0; e < 8; e++) o[t][e] *= alpha[e];
      v16bf vf = ldB(Vbase + (size_t)(t * 16 + l15) * SEQ + k0 + hl * 16);
      o[t] = wmma_bf16(pf, vf, o[t]);
    }
  }

  // ---- finalize: O /= l, store bf16 to (B,S,DIM)
#pragma unroll
  for (int t = 0; t < 8; t++) {
#pragma unroll
    for (int e = 0; e < 8; e++) {
      const float val = o[t][e] / lrow[e];
      const int srow = q0 + e + hl * 8;
      attn[((size_t)(b * SEQ + srow)) * DIM + h * HD + t * 16 + l15] = (__bf16)val;
    }
  }
}

// ------------------------------------------------------------------ launch --
extern "C" void kernel_launch(void* const* d_in, const int* in_sizes, int n_in,
                              void* d_out, int out_size, void* d_ws, size_t ws_size,
                              hipStream_t stream) {
  const float* x    = (const float*)d_in[0];
  const float* cosp = (const float*)d_in[2];
  const float* sinp = (const float*)d_in[3];
  const float* wq   = (const float*)d_in[5];
  const float* wk   = (const float*)d_in[6];
  const float* wv   = (const float*)d_in[7];
  const float* wo   = (const float*)d_in[8];

  __bf16* ws = (__bf16*)d_ws;
  const size_t E = (size_t)DIM * DIM;     // 16,777,216 elems (x is same size: B*S*DIM)
  __bf16* xb   = ws;
  __bf16* wqb  = ws + 1 * E;
  __bf16* wkb  = ws + 2 * E;
  __bf16* wvb  = ws + 3 * E;
  __bf16* wob  = ws + 4 * E;
  __bf16* Qbuf = ws + 5 * E;
  __bf16* Kbuf = ws + 6 * E;
  __bf16* Vbuf = ws + 7 * E;              // (B,H,HD,S) transposed
  __bf16* attn = xb;                      // reuse: x no longer needed after QKV GEMMs

  const dim3 cb(256), cg(4096);
  cvt_f32_bf16<<<cg, cb, 0, stream>>>(x,  xb,  (int)E);
  cvt_f32_bf16<<<cg, cb, 0, stream>>>(wq, wqb, (int)E);
  cvt_f32_bf16<<<cg, cb, 0, stream>>>(wk, wkb, (int)E);
  cvt_f32_bf16<<<cg, cb, 0, stream>>>(wv, wvb, (int)E);
  cvt_f32_bf16<<<cg, cb, 0, stream>>>(wo, wob, (int)E);

  const dim3 gg(DIM / 256, (BATCH * SEQ) / 128), gb(256);
  gemm_bf16<1><<<gg, gb, 0, stream>>>(xb, wqb, Qbuf, cosp, sinp);   // Q + RoPE
  gemm_bf16<1><<<gg, gb, 0, stream>>>(xb, wkb, Kbuf, cosp, sinp);   // K + RoPE
  gemm_bf16<0><<<gg, gb, 0, stream>>>(xb, wvb, Vbuf, nullptr, nullptr); // V transposed

  attn_flash<<<dim3(SEQ / 128, BATCH * NH), 256, 0, stream>>>(Qbuf, Kbuf, Vbuf, attn);

  gemm_bf16<2><<<gg, gb, 0, stream>>>(attn, wob, d_out, nullptr, nullptr); // out proj
}
